// UnifiedSARRenderer_27264452395686
// MI455X (gfx1250) — compile-verified, matrix operand order checked by reading, not run on hardware
//
#include <hip/hip_runtime.h>
#include <hip/hip_bf16.h>

typedef __attribute__((ext_vector_type(16))) _Float16 v16h;
typedef __attribute__((ext_vector_type(8)))  float    v8f;
typedef unsigned int u32x4 __attribute__((ext_vector_type(4)));
typedef int          i32x8 __attribute__((ext_vector_type(8)));
typedef int          i32x4 __attribute__((ext_vector_type(4)));

#define N_GAUSS 1024
#define NPAR    (6 * N_GAUSS)     // 6 float arrays -> 24 KB staged per block
#define NR 512
#define NA 512

// ---- compile-time geometry constants (BETA = asin(sin(30deg)) = pi/6) ----
#define CB   0.8660254037844387f          // cos(beta)
#define SB   0.5f                          // sin(beta)
#define RPY  (-2886.7513459481287f)        // radar pos y = -ALT*tan(beta)
#define RPZ  5000.0f                       // radar pos z = ALT
#define RCON 5773.502691896258f            // RC = ALT/cos(beta)  (rho_r = 1)
#define NLH  (-0.7213475204444817f)        // -0.5*log2(e): folded into coeffs

#if defined(__HIP_DEVICE_COMPILE__) && defined(__has_builtin)
#  if __has_builtin(__builtin_amdgcn_tensor_load_to_lds)
#    define HAVE_TDM 1
#  endif
#  if __has_builtin(__builtin_amdgcn_exp2f)
#    define FAST_EXP2(x) __builtin_amdgcn_exp2f(x)   // raw v_exp_f32 (2^x)
#  endif
#endif
#ifndef FAST_EXP2
#  define FAST_EXP2(x) exp2f(x)   // host pass / fallback (HIP device overload)
#endif

// =====================================================================
// Kernel 1: per-Gaussian preprocessing.
//   ws layout: rc[1024] cc[1024] iaL[1024] ibL[1024] idL[1024] w[1024]
//   iaL/ibL/idL are pre-scaled by -0.5*log2(e) (cross term also by 2) so
//   the splat loop feeds v_exp_f32 (2^x) with no pre-multiply or guards.
// =====================================================================
__global__ void sar_prep_kernel(const float* __restrict__ pos,
                                const float* __restrict__ cov,
                                const float* __restrict__ inten,
                                float* __restrict__ ws)
{
    int n = blockIdx.x * blockDim.x + threadIdx.x;
    if (n >= N_GAUSS) return;

    float p0 = pos[3 * n + 0];
    float p1 = pos[3 * n + 1] - RPY;
    float p2 = pos[3 * n + 2] - RPZ;
    float Xr =  p0;
    float Yr = -CB * p1 - SB * p2;
    float Zr =  SB * p1 - CB * p2;

    const float R[3][3] = { {1.f, 0.f, 0.f},
                            {0.f, -CB, -SB},
                            {0.f,  SB, -CB} };
    float C[3][3];
#pragma unroll
    for (int i = 0; i < 3; ++i)
#pragma unroll
        for (int j = 0; j < 3; ++j)
            C[i][j] = cov[9 * n + 3 * i + j];

    float M[3][3];   // M = C * R^T
#pragma unroll
    for (int i = 0; i < 3; ++i)
#pragma unroll
        for (int j = 0; j < 3; ++j)
            M[i][j] = C[i][0] * R[j][0] + C[i][1] * R[j][1] + C[i][2] * R[j][2];

    float cr00 = R[0][0]*M[0][0] + R[0][1]*M[1][0] + R[0][2]*M[2][0];
    float cr01 = R[0][0]*M[0][1] + R[0][1]*M[1][1] + R[0][2]*M[2][1];
    float cr02 = R[0][0]*M[0][2] + R[0][1]*M[1][2] + R[0][2]*M[2][2];
    float cr11 = R[1][0]*M[0][1] + R[1][1]*M[1][1] + R[1][2]*M[2][1];
    float cr12 = R[1][0]*M[0][2] + R[1][1]*M[1][2] + R[1][2]*M[2][2];
    float cr22 = R[2][0]*M[0][2] + R[2][1]*M[1][2] + R[2][2]*M[2][2];

    float Rmin = sqrtf(Yr * Yr + Zr * Zr + 1e-12f);
    float rc   = Rmin + (NR * 0.5f) - RCON;     // rho_r = 1
    float cc   = Xr   + (NA * 0.5f);            // rho_a = 1

    float j1 = Yr / Rmin;
    float j2 = Zr / Rmin;

    float a = j1 * j1 * cr11 + 2.0f * j1 * j2 * cr12 + j2 * j2 * cr22 + 1e-4f;
    float b = j1 * cr01 + j2 * cr02;
    float d = cr00 + 1e-4f;

    float det = a * d - b * b;
    float inv = 1.0f / det;

    ws[0 * N_GAUSS + n] = rc;
    ws[1 * N_GAUSS + n] = cc;
    ws[2 * N_GAUSS + n] = NLH * d * inv;                 // -0.5*log2e * ia
    ws[3 * N_GAUSS + n] = (-2.0f * NLH) * b * inv;       // -0.5*log2e * 2*ib
    ws[4 * N_GAUSS + n] = NLH * a * inv;                 // -0.5*log2e * id
    ws[5 * N_GAUSS + n] = inten[n] * (0.15915494309189535f * inv); // /(2*pi*det)
}

// =====================================================================
// Kernel 2: splat. 256 threads = 8 waves; wave w owns row blockIdx.y*8+w,
// cols [blockIdx.x*16,+16). Params DMA'd into LDS via the Tensor Data
// Mover (one 2D descriptor, 24 KB), then the K-loop runs 32-Gaussian
// chunks: each lane builds 16 f16 2^x values (B frag, K x N = 32 x 16)
// and a broadcast-weight A frag; v_wmma_f32_16x16x32_f16 accumulates the
// weighted n-sum in f32 (full-precision over all 1024 terms).
// =====================================================================
__global__ void __launch_bounds__(256)
sar_splat_kernel(const float* __restrict__ ws, float* __restrict__ out)
{
    __shared__ float s_par[NPAR];   // rc | cc | iaL | ibL | idL | w

    const int tid = threadIdx.x;

#if HAVE_TDM
    if ((tid >> 5) == 0) {  // wave 0 issues the TDM DMA (EXEC-independent)
        unsigned long long ga = (unsigned long long)(uintptr_t)ws;
        unsigned lds_off = (unsigned)(uintptr_t)s_par;  // low 32b = LDS offset
        // D# group0: count=1 | lds_addr | global_addr[56:0] | type=2
        u32x4 g0 = { 1u,
                     lds_off,
                     (unsigned)ga,
                     (unsigned)((ga >> 32) & 0x01FFFFFFull) | (2u << 30) };
        // D# group1: data_size=4B; tensor 6144x1, tile 6144x1, stride 6144
        i32x8 g1 = { 0x00020000,                         // data_size=2 (4B)
                     (int)((NPAR & 0xFFFF) << 16),       // tensor_dim0 lo
                     (int)((NPAR >> 16) | (1 << 16)),    // dim0 hi | tensor_dim1=1
                     (int)((NPAR & 0xFFFF) << 16),       // dim1 hi=0 | tile_dim0
                     1,                                  // tile_dim1=1, tile_dim2=0
                     NPAR,                               // tensor_dim0_stride lo
                     0, 0 };
        i32x4 gz = { 0, 0, 0, 0 };                        // groups 2/3 unused (2D)
#if __clang_major__ >= 23
        i32x8 gz8 = { 0, 0, 0, 0, 0, 0, 0, 0 };
        __builtin_amdgcn_tensor_load_to_lds(g0, g1, gz, gz, gz8, 0);
#else
        __builtin_amdgcn_tensor_load_to_lds(g0, g1, gz, gz, 0);
#endif
        __builtin_amdgcn_s_wait_tensorcnt(0);
    }
#else
    __builtin_prefetch(ws, 0, 1);
#pragma unroll
    for (int i = 0; i < NPAR / 256; ++i) {
        int g = tid + 256 * i;
        s_par[g] = ws[g];
    }
#endif
    __syncthreads();

    const float* s_rc = s_par;
    const float* s_cc = s_par + 1 * N_GAUSS;
    const float* s_ia = s_par + 2 * N_GAUSS;
    const float* s_ib = s_par + 3 * N_GAUSS;
    const float* s_id = s_par + 4 * N_GAUSS;
    const float* s_w  = s_par + 5 * N_GAUSS;

    const int lane = tid & 31;
    const int wave = tid >> 5;
    const int row  = blockIdx.y * 8 + wave;           // image row (0..511)
    const int col  = blockIdx.x * 16 + (lane & 15);   // image col (0..511)
    const int kofs = (lane >> 4) ? 8 : 0;             // K striping per lane half

    const float fr = (float)row;
    const float fc = (float)col;

    v8f acc = {};   // f32 accumulator; every M row identical by construction

    for (int chunk = 0; chunk < N_GAUSS; chunk += 32) {
        v16h afrag;   // A[m,k] = w[k]  (broadcast along M)
        v16h bfrag;   // B[k,p] = 2^(iaL*dr^2 + ibL*dr*dc + idL*dc^2)
#pragma unroll
        for (int e = 0; e < 16; ++e) {
            // 16-bit matrix VGPR striping: lanes 0-15 hold K = {0..7,16..23},
            // lanes 16-31 hold +8, two K per VGPR.
            int K = chunk + kofs + ((e < 8) ? e : (e + 8));
            float dr = fr - s_rc[K];
            float dc = fc - s_cc[K];
            float m  = s_ia[K] * (dr * dr)
                     + s_ib[K] * (dr * dc)
                     + s_id[K] * (dc * dc);
            afrag[e] = (_Float16)s_w[K];
            bfrag[e] = (_Float16)FAST_EXP2(m);   // single raw v_exp_f32
        }
        acc = __builtin_amdgcn_wmma_f32_16x16x32_f16(
                  false, afrag, false, bfrag,
                  (short)0, acc, false, false);
    }

    if (lane < 16) {
        out[row * NA + col] = acc[0];
    }
}

extern "C" void kernel_launch(void* const* d_in, const int* in_sizes, int n_in,
                              void* d_out, int out_size, void* d_ws, size_t ws_size,
                              hipStream_t stream)
{
    const float* positions   = (const float*)d_in[0];  // (1024, 3)
    const float* cov_world   = (const float*)d_in[1];  // (1024, 3, 3)
    const float* intensities = (const float*)d_in[2];  // (1024,)
    float* out = (float*)d_out;                        // (512, 512)
    float* ws  = (float*)d_ws;                         // 6 * 1024 floats

    sar_prep_kernel<<<(N_GAUSS + 255) / 256, 256, 0, stream>>>(
        positions, cov_world, intensities, ws);

    dim3 grid(NA / 16, NR / 8);   // 32 x 64 blocks
    sar_splat_kernel<<<grid, 256, 0, stream>>>(ws, out);
}
